// TemporalMemoryNetwork_31250182045986
// MI455X (gfx1250) — matrix operand order repaired and checked by reading, not compiled, not run on hardware
//
#include <hip/hip_runtime.h>

typedef unsigned short ushort_t;
typedef unsigned int   uint_t;

typedef __attribute__((ext_vector_type(16))) __bf16 bf16x16;
typedef __attribute__((ext_vector_type(8)))  float  f32x8;

#define INPUT_DIM 1024
#define MEM_DIM   512
#define NKEYS     4096
#define TOPK      8
#define ROWS      32      // rows per workgroup
#define QPITCH    520     // bf16 pitch for q / mem_out LDS tiles
#define NCHUNK    64      // score columns per sweep iteration
#define SPITCH    68      // float pitch for score staging (float4-aligned, bank-spread)

union ABf { bf16x16 v; uint4 q[2]; };
union BF8 { __bf16 h[8]; uint4 u; };

__device__ __forceinline__ f32x8 wmma_bf16(bf16x16 a, bf16x16 b, f32x8 c) {
    // (neg_a, A, neg_b, B, c_mod, C, reuse_a, reuse_b)
    return __builtin_amdgcn_wmma_f32_16x16x32_bf16(false, a, false, b, (short)0, c, false, false);
}

// Static-index ascending top-8 insert (tv[0] = current min). Call only when v > tv[0].
__device__ __forceinline__ void top8_shift_insert(float* tv, int* ti, float v, int gi) {
    bool prev_gt = true;                       // v > tv[0] guaranteed by caller
    #pragma unroll
    for (int j = 0; j < TOPK - 1; ++j) {
        const bool gt_next = v > tv[j + 1];
        const float cv0 = tv[j + 1]; const int ci0 = ti[j + 1];
        tv[j] = gt_next ? cv0 : (prev_gt ? v : tv[j]);
        ti[j] = gt_next ? ci0 : (prev_gt ? gi : ti[j]);
        prev_gt = gt_next;
    }
    tv[TOPK - 1] = prev_gt ? v  : tv[TOPK - 1];
    ti[TOPK - 1] = prev_gt ? gi : ti[TOPK - 1];
}

// ---------------- prep kernels ----------------
// keys: straight f32 -> bf16, 8 elems/thread, b128 in / b128 out
__global__ void tmn_cvt_keys(const float4* __restrict__ in, uint4* __restrict__ out) {
    const int i = blockIdx.x * 256 + threadIdx.x;
    const float4 a = in[i * 2], b = in[i * 2 + 1];
    BF8 o;
    o.h[0] = (__bf16)a.x; o.h[1] = (__bf16)a.y; o.h[2] = (__bf16)a.z; o.h[3] = (__bf16)a.w;
    o.h[4] = (__bf16)b.x; o.h[5] = (__bf16)b.y; o.h[6] = (__bf16)b.z; o.h[7] = (__bf16)b.w;
    out[i] = o.u;
}
// LDS-tiled transpose: in [K][N] f32 (k-major) -> out [N][K] bf16 (n-major)
__global__ void tmn_transpose_bf16(const float* __restrict__ in, __bf16* __restrict__ out,
                                   int K, int N) {
    __shared__ float t[32][33];
    const int nb = N >> 5;
    const int n0 = (blockIdx.x % nb) * 32;
    const int k0 = (blockIdx.x / nb) * 32;
    const int lx = threadIdx.x & 31, ly = threadIdx.x >> 5;    // 256 thr = 32x8
    #pragma unroll
    for (int r = 0; r < 32; r += 8) t[ly + r][lx] = in[(k0 + ly + r) * N + n0 + lx];
    __syncthreads();
    #pragma unroll
    for (int r = 0; r < 32; r += 8) out[(size_t)(n0 + ly + r) * K + k0 + lx] = (__bf16)t[lx][ly + r];
}

// ---------------- fused main kernel: 32 rows per workgroup, 8 waves ----------------
__global__ __launch_bounds__(256) void tmn_fused(
    const float* __restrict__ query,    // [32768][1024]
    const __bf16* __restrict__ keysb,   // [4096][512] bf16
    const float* __restrict__ values,   // [4096][512] f32
    const __bf16* __restrict__ wqT,     // [512][1024] bf16 (N-major)
    const __bf16* __restrict__ woT,     // [1024][512] bf16 (N-major)
    const float* __restrict__ bq,       // [512]
    const float* __restrict__ bo,       // [1024]
    float* __restrict__ out)            // [32768][1024]
{
    __shared__ __align__(16) unsigned char smem[52736];
    __bf16* q_mo = (__bf16*)smem;                    // 32*520 bf16 : q tile, reused as mem_out
    float*  scb  = (float*)(smem + 33280);           // 2 x 32*68 f32 score staging (double-buffer)
    float*  cvv  = (float*)(smem + 33280);           // overlays scb: 32*64 candidate values
    int*    cii  = (int*)  (smem + 33280 + 8704);    // overlays scb: 32*64 candidate indices
    float*  w8   = (float*)(smem + 50688);           // 32*8 softmax weights
    int*    i8s  = (int*)  (smem + 51712);           // 32*8 selected indices

    const int tid  = threadIdx.x;
    const int w    = tid >> 5;
    const int lane = tid & 31;
    const int lm   = lane & 15;
    const int lhi  = (lane & 16) ? 1 : 0;
    const int row0 = blockIdx.x * ROWS;

    const int mtile = w & 1;              // 0..1 : which 16-row half
    const int ngrp  = w >> 1;             // 0..3 : N-group

    f32x8 zero = {0.f,0.f,0.f,0.f,0.f,0.f,0.f,0.f};

    // ========== Phase 1: q = query @ wq + bq  -> q_mo (bf16) ==========
    {
        f32x8 acc[8];
        #pragma unroll
        for (int j = 0; j < 8; ++j) acc[j] = zero;
        const int gm = row0 + mtile * 16 + lm;
        for (int ks = 0; ks < INPUT_DIM / 32; ++ks) {
            const int k0 = ks * 32;
            ABf a;
            const float* ap = query + gm * INPUT_DIM + k0 + (lhi ? 8 : 0);
            float fl[16];
            *(float4*)&fl[0]  = ((const float4*)ap)[0];
            *(float4*)&fl[4]  = ((const float4*)ap)[1];
            *(float4*)&fl[8]  = ((const float4*)(ap + 16))[0];
            *(float4*)&fl[12] = ((const float4*)(ap + 16))[1];
            #pragma unroll
            for (int i = 0; i < 16; ++i) a.v[i] = (__bf16)fl[i];   // native v_cvt path
            #pragma unroll
            for (int j = 0; j < 8; ++j) {
                const int n0 = (ngrp * 8 + j) * 16;
                ABf b;
                const uint4* bp = (const uint4*)(wqT + (n0 + lm) * INPUT_DIM + k0 + (lhi ? 8 : 0));
                b.q[0] = bp[0]; b.q[1] = bp[2];
                acc[j] = wmma_bf16(a.v, b.v, acc[j]);
            }
        }
        #pragma unroll
        for (int j = 0; j < 8; ++j) {
            const int col = (ngrp * 8 + j) * 16 + lm;
            const float bqv = bq[col];
            #pragma unroll
            for (int e = 0; e < 8; ++e) {
                const int r = mtile * 16 + e + lhi * 8;
                q_mo[r * QPITCH + col] = (__bf16)(acc[j][e] + bqv);
            }
        }
    }
    __syncthreads();

    // ========== Phase 2: sweep scores = q . keys, running per-row top-8 ==========
    float tv[TOPK]; int ti[TOPK];                    // ascending (tv[0] = min)
    #pragma unroll
    for (int i = 0; i < TOPK; ++i) { tv[i] = -3.0e38f; ti[i] = 0; }
    const int trow  = tid >> 3;                      // 0..31 : row this thread scans
    const int tslot = tid & 7;                       // 0..7  : contiguous 8-col subset

    for (int c = 0; c < NKEYS / NCHUNK; ++c) {
        const int nbase = c * NCHUNK;
        float* sc = scb + (c & 1) * (ROWS * SPITCH);
        f32x8 acc = zero;
        const int n0 = nbase + ngrp * 16;
        #pragma unroll
        for (int ks = 0; ks < MEM_DIM / 32; ++ks) {
            const int k0 = ks * 32;
            ABf a;
            const __bf16* apq = q_mo + (mtile * 16 + lm) * QPITCH + k0 + (lhi ? 8 : 0);
            a.q[0] = *(const uint4*)apq;
            a.q[1] = *(const uint4*)(apq + 16);
            ABf b;
            const uint4* bp = (const uint4*)(keysb + (n0 + lm) * MEM_DIM + k0 + (lhi ? 8 : 0));
            b.q[0] = bp[0]; b.q[1] = bp[2];
            acc = wmma_bf16(a.v, b.v, acc);
        }
        #pragma unroll
        for (int e = 0; e < 8; ++e) {
            const int r = mtile * 16 + e + lhi * 8;
            sc[r * SPITCH + ngrp * 16 + lm] = acc[e];
        }
        __syncthreads();
        // scan: two conflict-free b128 LDS loads per thread
        const float* sp = sc + trow * SPITCH + tslot * 8;
        const float4 v0 = ((const float4*)sp)[0];
        const float4 v1 = ((const float4*)sp)[1];
        const int gi0 = nbase + tslot * 8;
        if (v0.x > tv[0]) top8_shift_insert(tv, ti, v0.x, gi0 + 0);
        if (v0.y > tv[0]) top8_shift_insert(tv, ti, v0.y, gi0 + 1);
        if (v0.z > tv[0]) top8_shift_insert(tv, ti, v0.z, gi0 + 2);
        if (v0.w > tv[0]) top8_shift_insert(tv, ti, v0.w, gi0 + 3);
        if (v1.x > tv[0]) top8_shift_insert(tv, ti, v1.x, gi0 + 4);
        if (v1.y > tv[0]) top8_shift_insert(tv, ti, v1.y, gi0 + 5);
        if (v1.z > tv[0]) top8_shift_insert(tv, ti, v1.z, gi0 + 6);
        if (v1.w > tv[0]) top8_shift_insert(tv, ti, v1.w, gi0 + 7);
        // no trailing barrier: next chunk writes the other buffer
    }
    __syncthreads();   // protect candidate region (overlays score buffers)

    // ========== Phase 3: merge 8 partial top-8s per row, softmax ==========
    #pragma unroll
    for (int i = 0; i < TOPK; ++i) {
        cvv[trow * 64 + tslot * 8 + i] = tv[i];
        cii[trow * 64 + tslot * 8 + i] = ti[i];
    }
    __syncthreads();
    if (tid < ROWS) {
        float mv[TOPK]; int mi[TOPK];
        #pragma unroll
        for (int i = 0; i < TOPK; ++i) { mv[i] = -3.0e38f; mi[i] = 0; }
        for (int j = 0; j < 64; ++j) {
            const float v = cvv[tid * 64 + j];
            if (v > mv[0]) top8_shift_insert(mv, mi, v, cii[tid * 64 + j]);
        }
        const float m = mv[TOPK - 1];
        float s = 0.f;
        #pragma unroll
        for (int i = 0; i < TOPK; ++i) s += __expf(mv[i] - m);
        const float inv = 1.0f / s;
        #pragma unroll
        for (int i = 0; i < TOPK; ++i) { w8[tid * 8 + i] = __expf(mv[i] - m) * inv; i8s[tid * 8 + i] = mi[i]; }
    }
    __syncthreads();

    // ========== Phase 4: mem_out = sum_i w_i * values[idx_i]  -> q_mo (bf16, reuse) ==========
    {
        float wgt[TOPK]; int idx[TOPK];
        #pragma unroll
        for (int i = 0; i < TOPK; ++i) { wgt[i] = w8[trow * 8 + i]; idx[i] = i8s[trow * 8 + i]; }
        const int cb = tslot * 4;
        for (int jj = 0; jj < 16; ++jj) {
            const int col = cb + jj * 32;
            float4 s4 = {0.f, 0.f, 0.f, 0.f};
            #pragma unroll
            for (int i = 0; i < TOPK; ++i) {
                const float4 v4 = *(const float4*)(values + idx[i] * MEM_DIM + col);
                s4.x += wgt[i] * v4.x; s4.y += wgt[i] * v4.y;
                s4.z += wgt[i] * v4.z; s4.w += wgt[i] * v4.w;
            }
            __bf16* dst = q_mo + trow * QPITCH + col;
            dst[0] = (__bf16)s4.x; dst[1] = (__bf16)s4.y;
            dst[2] = (__bf16)s4.z; dst[3] = (__bf16)s4.w;
        }
    }
    __syncthreads();

    // ========== Phase 5: out = mem_out @ wo + bo + query ==========
    for (int p = 0; p < 2; ++p) {
        f32x8 acc[8];
        #pragma unroll
        for (int j = 0; j < 8; ++j) acc[j] = zero;
        for (int ks = 0; ks < MEM_DIM / 32; ++ks) {
            const int k0 = ks * 32;
            ABf a;
            const __bf16* apq = q_mo + (mtile * 16 + lm) * QPITCH + k0 + (lhi ? 8 : 0);
            a.q[0] = *(const uint4*)apq;
            a.q[1] = *(const uint4*)(apq + 16);
            #pragma unroll
            for (int j = 0; j < 8; ++j) {
                const int nt = ngrp * 16 + p * 8 + j;
                ABf b;
                const uint4* bp = (const uint4*)(woT + (nt * 16 + lm) * MEM_DIM + k0 + (lhi ? 8 : 0));
                b.q[0] = bp[0]; b.q[1] = bp[2];
                acc[j] = wmma_bf16(a.v, b.v, acc[j]);
            }
        }
        #pragma unroll
        for (int j = 0; j < 8; ++j) {
            const int col = (ngrp * 16 + p * 8 + j) * 16 + lm;
            const float bov = bo[col];
            #pragma unroll
            for (int e = 0; e < 8; ++e) {
                const int r = row0 + mtile * 16 + e + lhi * 8;
                out[r * INPUT_DIM + col] = acc[j][e] + bov + query[r * INPUT_DIM + col];
            }
        }
    }
}

extern "C" void kernel_launch(void* const* d_in, const int* in_sizes, int n_in,
                              void* d_out, int out_size, void* d_ws, size_t ws_size,
                              hipStream_t stream) {
    const float* query = (const float*)d_in[0];
    const float* mkeys = (const float*)d_in[1];
    const float* mvals = (const float*)d_in[2];
    const float* wq    = (const float*)d_in[3];
    const float* bq    = (const float*)d_in[4];
    const float* wo    = (const float*)d_in[5];
    const float* bo    = (const float*)d_in[6];
    float* out = (float*)d_out;

    __bf16* keysb = (__bf16*)d_ws;                           // 4 MiB
    __bf16* wqT   = keysb + (size_t)NKEYS * MEM_DIM;         // 1 MiB
    __bf16* woT   = wqT + (size_t)MEM_DIM * INPUT_DIM;       // 1 MiB

    tmn_cvt_keys<<<(NKEYS * MEM_DIM) / (256 * 8), 256, 0, stream>>>(
        (const float4*)mkeys, (uint4*)keysb);
    tmn_transpose_bf16<<<(MEM_DIM / 32) * (INPUT_DIM / 32), 256, 0, stream>>>(
        wq, wqT, INPUT_DIM, MEM_DIM);   // in [1024][512] -> out [512][1024]
    tmn_transpose_bf16<<<(INPUT_DIM / 32) * (MEM_DIM / 32), 256, 0, stream>>>(
        wo, woT, MEM_DIM, INPUT_DIM);   // in [512][1024] -> out [1024][512]

    const int total_rows = 8 * 4096;
    tmn_fused<<<total_rows / ROWS, 256, 0, stream>>>(query, keysb, mvals, wqT, woT, bq, bo, out);
}
